// SimpleConvGRU_16776142259107
// MI455X (gfx1250) — matrix operand order repaired and checked by reading, not compiled, not run on hardware
//
#include <hip/hip_runtime.h>
#include <stdint.h>

#define BB 16
#define TT 28
#define HH 32
#define WW 32
#define FF 64
#define BTHW (BB*TT*HH*WW)   /* 458752 */
#define BHW  (BB*HH*WW)      /* 16384  */
#define PH 34
#define PW 34
#define PHW (PH*PW)          /* 1156 padded pixels per image */

typedef __attribute__((ext_vector_type(16))) __bf16 v16bf;
typedef __attribute__((ext_vector_type(8)))  float  v8f;

union FragBF { v16bf v; uint4 q[2]; };

__device__ __forceinline__ long pad_idx(int img, int y, int x) {
  return (long)img * PHW + (long)(y + 1) * PW + (x + 1);
}

// ---------------------------------------------------------------- zero words
__global__ void k_zero32(uint32_t* __restrict__ p, long n) {
  long i = (long)blockIdx.x * blockDim.x + threadIdx.x;
  if (i < n) p[i] = 0u;
}

// ---------------------------------------------------------------- zero h state (f32 dense + padded bf16)
__global__ void k_zero_h(float* __restrict__ hf, uint32_t* __restrict__ hbw,
                         long nf, long nbw) {
  long i = (long)blockIdx.x * blockDim.x + threadIdx.x;
  if (i < nf)  hf[i]  = 0.f;
  if (i < nbw) hbw[i] = 0u;
}

// ---------------------------------------------------------------- cast+pad input x: (img,32,32,24)f32 -> (img,34,34,32)bf16 interior
__global__ void k_cast_pad_x(const float* __restrict__ s, __bf16* __restrict__ d) {
  long p = (long)blockIdx.x * blockDim.x + threadIdx.x;
  if (p >= (long)BTHW) return;
  int img = (int)(p >> 10), rem = (int)(p & 1023), y = rem >> 5, x = rem & 31;
  long db = pad_idx(img, y, x) * 32;
  long sb = p * 24;
#pragma unroll
  for (int c = 0; c < 24; ++c) d[db + c] = (__bf16)s[sb + c];
}

// ---------------------------------------------------------------- pack weights into WMMA B-fragment order
// w layout (ky,kx,Cin,Cout). Packed: [tap][kblock][ntile][lane][16 bf16]
// B 32x16 bf16 fragment (col n=lane&15): lanes 0-15 hold K=0..7,16..23; lanes 16-31 hold K=8..15,24..31.
__global__ void k_pack_w(const float* __restrict__ w, __bf16* __restrict__ pk,
                         int Cin, int Cout, int KB, int NT) {
  long id = (long)blockIdx.x * blockDim.x + threadIdx.x;
  long total = (long)9 * KB * NT * 32 * 8;
  if (id >= total) return;
  int i    = (int)(id & 7);  long r = id >> 3;
  int lane = (int)(r & 31);  r >>= 5;
  int nt   = (int)(r % NT);  r /= NT;
  int kbi  = (int)(r % KB);  r /= KB;
  int t    = (int)r;
  int n  = nt * 16 + (lane & 15);
  int k0 = kbi * 32 + ((lane >> 4) * 8) + 2 * (i & 3) + (i >> 2) * 16;
  float a0 = 0.f, a1 = 0.f;
  if (n < Cout) {
    if (k0     < Cin) a0 = w[((long)t * Cin + k0    ) * Cout + n];
    if (k0 + 1 < Cin) a1 = w[((long)t * Cin + k0 + 1) * Cout + n];
  }
  long base = ((((long)t * KB + kbi) * NT + nt) * 32 + lane) * 16 + 2 * i;
  pk[base]     = (__bf16)a0;
  pk[base + 1] = (__bf16)a1;
}

// ---------------------------------------------------------------- implicit-GEMM 3x3 conv on padded input via WMMA bf16
// grid: (mtiles/8, NG). One wave: 32(M) x 16*NTPW(N). No guards anywhere; EXEC all-ones.
// LDSB: stage packed-B for this block's n-group into LDS via global_load_async_to_lds_b128.
template<int KB, int CSTR, int NTPW, bool OUTBF, bool LDSB>
__global__ void __launch_bounds__(256)
k_conv3x3_wmma(const __bf16* __restrict__ act,
               const __bf16* __restrict__ wpk, int NTw, int ntw_base,
               const float* __restrict__ bias,
               void* __restrict__ outp, int out_cstride) {
  constexpr int SMEMB = LDSB ? 9 * KB * NTPW * 1024 : 16;
  __shared__ __align__(16) char smemB[SMEMB];

  const int lane = threadIdx.x & 31;
  const int mt   = blockIdx.x * 8 + (threadIdx.x >> 5);
  const int nt0  = blockIdx.y * NTPW;               // block-uniform n-group

  if (LDSB) {
    // cooperative async stage of all B fragments for this n-group (16B chunks)
    const int chunks = 9 * KB * NTPW * 64;
    for (int c = threadIdx.x; c < chunks; c += 256) {
      int half = c & 1;  int r = c >> 1;
      int ln = r & 31;   r >>= 5;
      int j  = r % NTPW; r /= NTPW;
      int kb = r % KB;   int t = r / KB;
      long goff = ((((long)t * KB + kb) * NTw) + ntw_base + nt0 + j) * 512 + ln * 16 + half * 8;
      unsigned lbase = (unsigned)(uintptr_t)(&smemB[0]) + (unsigned)(c * 16);
      uint64_t ga = (uint64_t)(uintptr_t)(wpk + goff);
      asm volatile("global_load_async_to_lds_b128 %0, %1, off"
                   :: "v"(lbase), "v"(ga) : "memory");
    }
    asm volatile("s_wait_asynccnt 0" ::: "memory");
    __syncthreads();
  }

  // A fragment lane mapping (16-bit A 16x32)
  const int mrow  = lane & 15;
  const int khalf = (lane >> 4) << 3;
  const int p0  = mt * 32 + mrow;                   // second A tile at p0+16 (same padded row)
  const int img = p0 >> 10;
  const int rem = p0 & 1023;
  const int y   = rem >> 5;
  const int x0  = rem & 31;                         // 0..15
  const __bf16* abase = act + ((long)(img * PH + y) * PW + x0) * CSTR;

  v8f acc[2][NTPW] = {};

#pragma unroll
  for (int t = 0; t < 9; ++t) {
    const int toff = ((t / 3) * PW + (t % 3)) * CSTR;   // compile-time per tap
    const __bf16* arow0 = abase + toff;
    const __bf16* arow1 = arow0 + 16 * CSTR;
#pragma unroll
    for (int kb = 0; kb < KB; ++kb) {
      const int c0 = kb * 32 + khalf;
      const int c1 = c0 + 16;
      FragBF A0, A1;
      A0.q[0] = *(const uint4*)(arow0 + c0);
      A0.q[1] = *(const uint4*)(arow0 + c1);
      A1.q[0] = *(const uint4*)(arow1 + c0);
      A1.q[1] = *(const uint4*)(arow1 + c1);
      FragBF Bf[NTPW];
      if (LDSB) {
        const char* wl = smemB + (((t * KB + kb) * NTPW) * 32 + lane) * 32;
#pragma unroll
        for (int j = 0; j < NTPW; ++j) {
          Bf[j].q[0] = *(const uint4*)(wl + j * 1024);
          Bf[j].q[1] = *(const uint4*)(wl + j * 1024 + 16);
        }
      } else {
        const __bf16* wb = wpk + ((((long)t * KB + kb) * NTw) + ntw_base + nt0) * 512 + lane * 16;
#pragma unroll
        for (int j = 0; j < NTPW; ++j) {
          Bf[j].q[0] = *(const uint4*)(wb + j * 512);
          Bf[j].q[1] = *(const uint4*)(wb + j * 512 + 8);
        }
      }
#pragma unroll
      for (int j = 0; j < NTPW; ++j) {
        acc[0][j] = __builtin_amdgcn_wmma_f32_16x16x32_bf16(
            false, A0.v, false, Bf[j].v, (short)0, acc[0][j], false, false);
        acc[1][j] = __builtin_amdgcn_wmma_f32_16x16x32_bf16(
            false, A1.v, false, Bf[j].v, (short)0, acc[1][j], false, false);
      }
    }
  }

  // Epilogue. C/D 16x16 f32 layout: col n = lane&15, VGPR i -> row i + (lane>=16 ? 8 : 0)
  const int ncol = lane & 15;
  const int mro  = (lane >> 4) << 3;
#pragma unroll
  for (int half = 0; half < 2; ++half) {
#pragma unroll
    for (int j = 0; j < NTPW; ++j) {
      const int   co_loc = (nt0 + j) * 16 + ncol;
      const float bv = bias ? bias[(ntw_base + nt0 + j) * 16 + ncol] : 0.f;
#pragma unroll
      for (int i = 0; i < 8; ++i) {
        const int   pm = mt * 32 + half * 16 + mro + i;
        const float v  = acc[half][j][i] + bv;
        const long  oi = (long)pm * out_cstride + co_loc;
        if (OUTBF) ((__bf16*)outp)[oi] = (__bf16)v;
        else       ((float*)outp)[oi]  = v;
      }
    }
  }
}

template<int KB, int CSTR, int NTPW, bool OUTBF, bool LDSB>
static void launch_conv(const __bf16* act, const __bf16* wpk, int NTw, int ntw_base,
                        const float* bias, void* outp, int out_cstride,
                        int npix, int NTloc, hipStream_t stream) {
  dim3 grid(npix / 32 / 8, NTloc / NTPW, 1);
  k_conv3x3_wmma<KB, CSTR, NTPW, OUTBF, LDSB><<<grid, 256, 0, stream>>>(
      act, wpk, NTw, ntw_base, bias, outp, out_cstride);
}

// ---------------------------------------------------------------- GRU gate stage 1: z, r, r*h (rh written padded)
__global__ void k_gates1(const __bf16* __restrict__ xg, int time,
                         const float* __restrict__ ghzr,
                         const float* __restrict__ h,
                         float* __restrict__ zbuf, __bf16* __restrict__ rh) {
  long id = (long)blockIdx.x * blockDim.x + threadIdx.x;
  if (id >= (long)BHW * FF) return;
  long p = id >> 6; int f = (int)(id & 63);
  int  b = (int)(p >> 10);
  int yx = (int)(p & 1023);
  int yy = yx >> 5, xx = yx & 31;
  long gx = ((long)(b * TT + time) * 1024 + yx) * 192;
  float gz = (float)xg[gx + f]      + ghzr[p * 128 + f];
  float gr = (float)xg[gx + 64 + f] + ghzr[p * 128 + 64 + f];
  float z = 1.f / (1.f + __expf(-gz));
  float r = 1.f / (1.f + __expf(-gr));
  zbuf[id] = z;
  rh[pad_idx(b, yy, xx) * 64 + f] = (__bf16)(r * h[id]);
}

// ---------------------------------------------------------------- GRU gate stage 2: hh, h update, write padded h + concat
__global__ void k_gates2(const __bf16* __restrict__ xg, int time,
                         const float* __restrict__ ghh,
                         const float* __restrict__ zbuf,
                         float* __restrict__ h, __bf16* __restrict__ hb,
                         __bf16* __restrict__ xc, int choff, int layer) {
  long id = (long)blockIdx.x * blockDim.x + threadIdx.x;
  if (id >= (long)BHW * FF) return;
  long p = id >> 6; int f = (int)(id & 63);
  int  b = (int)(p >> 10);
  int yx = (int)(p & 1023);
  int yy = yx >> 5, xx = yx & 31;
  long gx = ((long)(b * TT + time) * 1024 + yx);
  float g  = (float)xg[gx * 192 + 128 + f] + ghh[id];
  float hh = tanhf(g);
  float z  = zbuf[id];
  float hn = z * h[id] + (1.f - z) * hh;
  h[id] = hn;
  hb[pad_idx(b, yy, xx) * 64 + f] = (__bf16)hn;
  long oc = pad_idx(b * TT + time, yy, xx) * 256;
  if (layer == 0) {
    xc[oc + 128 + choff + f] = (__bf16)hn;                               // x0 half of concat
  } else {
    xc[oc + choff + f] = (__bf16)((float)xc[oc + 128 + choff + f] + hn); // xr = x0 + l1out
  }
}

// ---------------------------------------------------------------- 1x1 head + relu
__global__ void k_out(const float* __restrict__ hm, const float* __restrict__ wo,
                      const float* __restrict__ bo, float* __restrict__ out) {
  long id = (long)blockIdx.x * blockDim.x + threadIdx.x;
  if (id >= (long)BTHW) return;
  float s = bo[0];
#pragma unroll
  for (int i = 0; i < 32; ++i) s += hm[id * 32 + i] * wo[i];
  out[id] = fmaxf(s, 0.f);
}

// ================================================================ host
extern "C" void kernel_launch(void* const* d_in, const int* in_sizes, int n_in,
                              void* d_out, int out_size, void* d_ws, size_t ws_size,
                              hipStream_t stream) {
  (void)in_sizes; (void)n_in; (void)out_size; (void)ws_size;
  const float* x      = (const float*)d_in[0];
  const float* wx0[2] = {(const float*)d_in[1],  (const float*)d_in[4]};
  const float* wh0[2] = {(const float*)d_in[2],  (const float*)d_in[5]};
  const float* b0[2]  = {(const float*)d_in[3],  (const float*)d_in[6]};
  const float* wx1[2] = {(const float*)d_in[7],  (const float*)d_in[10]};
  const float* wh1[2] = {(const float*)d_in[8],  (const float*)d_in[11]};
  const float* b1[2]  = {(const float*)d_in[9],  (const float*)d_in[12]};
  const float* wc1    = (const float*)d_in[13];
  const float* bc1    = (const float*)d_in[14];
  const float* wo     = (const float*)d_in[15];
  const float* bo     = (const float*)d_in[16];

  char* wsp = (char*)d_ws;
  auto take = [&](size_t bytes) -> void* {
    void* r = (void*)wsp;
    wsp += (bytes + 255) & ~(size_t)255;
    return r;
  };
  const long PIMG_T = (long)BB * TT;                 // 448 padded images (per-timestep)
  __bf16* x_bf = (__bf16*)take((size_t)PIMG_T * PHW * 32  * 2);  // padded, Cin padded to 32
  __bf16* xc   = (__bf16*)take((size_t)PIMG_T * PHW * 256 * 2);  // padded concat buffer
  __bf16* hb   = (__bf16*)take((size_t)BB * PHW * 64 * 2);       // padded h (bf16)
  __bf16* rh   = (__bf16*)take((size_t)BB * PHW * 64 * 2);       // padded r*h
  __bf16* xg   = (__bf16*)take((size_t)BTHW * 192 * 2);          // dense
  float*  hf   = (float*) take((size_t)BHW * FF * 4);
  float*  zb   = (float*) take((size_t)BHW * FF * 4);
  float*  ghzr = (float*) take((size_t)BHW * 128 * 4);
  float*  ghh  = (float*) take((size_t)BHW * 64  * 4);
  float*  hmid = (float*) take((size_t)BTHW * 32 * 4);
  __bf16* pk_wx0[2], *pk_wx1[2], *pk_wh0[2], *pk_wh1[2], *pk_wc1;
  for (int d = 0; d < 2; ++d) pk_wx0[d] = (__bf16*)take((size_t)9 * 1 * 12 * 1024);
  for (int d = 0; d < 2; ++d) pk_wx1[d] = (__bf16*)take((size_t)9 * 4 * 12 * 1024);
  for (int d = 0; d < 2; ++d) pk_wh0[d] = (__bf16*)take((size_t)9 * 2 * 12 * 1024);
  for (int d = 0; d < 2; ++d) pk_wh1[d] = (__bf16*)take((size_t)9 * 2 * 12 * 1024);
  pk_wc1 = (__bf16*)take((size_t)9 * 8 * 2 * 1024);

  auto zero = [&](void* p, size_t bytes) {
    long nw = (long)(bytes / 4);
    k_zero32<<<(int)((nw + 255) / 256), 256, 0, stream>>>((uint32_t*)p, nw);
  };
  auto pack = [&](const float* wsrc, __bf16* dst, int Cin, int Cout, int KB, int NT) {
    long n = (long)9 * KB * NT * 32 * 8;
    k_pack_w<<<(int)((n + 255) / 256), 256, 0, stream>>>(wsrc, dst, Cin, Cout, KB, NT);
  };

  // zero padded buffers once (borders + channel pad stay zero; interiors rewritten by producers)
  zero(x_bf, (size_t)PIMG_T * PHW * 32 * 2);
  zero(xc,   (size_t)PIMG_T * PHW * 256 * 2);
  zero(rh,   (size_t)BB * PHW * 64 * 2);

  { // cast + pad input activations to bf16 (24 -> 32 channels)
    k_cast_pad_x<<<(int)(((long)BTHW + 255) / 256), 256, 0, stream>>>(x, x_bf);
  }
  for (int d = 0; d < 2; ++d) {
    pack(wx0[d], pk_wx0[d], 24, 192, 1, 12);
    pack(wx1[d], pk_wx1[d], 128, 192, 4, 12);
    pack(wh0[d], pk_wh0[d], 64, 192, 2, 12);
    pack(wh1[d], pk_wh1[d], 64, 192, 2, 12);
  }
  pack(wc1, pk_wc1, 256, 32, 8, 2);

  const long nf  = (long)BHW * FF;                       // dense h floats
  const long nbw = (long)BB * PHW * 64 * 2 / 4;          // padded h bf16, in words
  const long nzh = (nf > nbw) ? nf : nbw;

  for (int layer = 0; layer < 2; ++layer) {
    for (int dir = 0; dir < 2; ++dir) {
      // all-timestep input-gate conv: xg = conv(x, wx) + b  (bf16 out, 192 ch)
      if (layer == 0)
        launch_conv<1, 32, 4, true, true>(x_bf, pk_wx0[dir], 12, 0, b0[dir],
                                          xg, 192, BTHW, 12, stream);
      else
        launch_conv<4, 256, 4, true, false>(xc + 128, pk_wx1[dir], 12, 0, b1[dir],
                                            xg, 192, BTHW, 12, stream);
      k_zero_h<<<(int)((nzh + 255) / 256), 256, 0, stream>>>(hf, (uint32_t*)hb, nf, nbw);
      const __bf16* pkh = (layer == 0) ? pk_wh0[dir] : pk_wh1[dir];
      for (int t = 0; t < TT; ++t) {
        const int time = dir ? (TT - 1 - t) : t;
        // gh_zr = conv(h, wh[:, :128])
        launch_conv<2, 64, 4, false, true>(hb, pkh, 12, 0, nullptr, ghzr, 128, BHW, 8, stream);
        { long n = (long)BHW * FF;
          k_gates1<<<(int)((n + 255) / 256), 256, 0, stream>>>(xg, time, ghzr, hf, zb, rh); }
        // gh_h = conv(r*h, wh[:, 128:])  (weight n-tiles 8..11)
        launch_conv<2, 64, 4, false, true>(rh, pkh, 12, 8, nullptr, ghh, 64, BHW, 4, stream);
        { long n = (long)BHW * FF;
          k_gates2<<<(int)((n + 255) / 256), 256, 0, stream>>>(xg, time, ghh, zb, hf, hb,
                                                              xc, dir * 64, layer); }
      }
    }
  }
  // conv1 over concat [xr, x0] (256 ch) -> 32, then 1x1 head + relu
  launch_conv<8, 256, 2, false, false>(xc, pk_wc1, 2, 0, bc1, hmid, 32, BTHW, 2, stream);
  k_out<<<(int)(((long)BTHW + 255) / 256), 256, 0, stream>>>(hmid, wo, bo, (float*)d_out);
}
